// StaticFusedMOE_4861902979823
// MI455X (gfx1250) — matrix operand order repaired and checked by reading, not compiled, not run on hardware
//
#include <hip/hip_runtime.h>
#include <math.h>

typedef __bf16 bf16;
typedef __bf16 bf8_t  __attribute__((ext_vector_type(8)));
typedef __bf16 v16bf  __attribute__((ext_vector_type(16)));
typedef float  v8f    __attribute__((ext_vector_type(8)));

#define B_TOK 4096
#define D_DIM 2048
#define E_NUM 8
#define I_DIM 2816
#define TWO_I 5632
#define ROWCAP 9216  // sum of 128-padded per-expert row counts <= 8192 + 8*127

#define SHUF16(lo, hi) __builtin_shufflevector((lo), (hi), 0,1,2,3,4,5,6,7,8,9,10,11,12,13,14,15)

static __device__ __forceinline__ bf8_t cvt8(float4 a, float4 b) {
  bf8_t r;
  r[0] = (bf16)a.x; r[1] = (bf16)a.y; r[2] = (bf16)a.z; r[3] = (bf16)a.w;
  r[4] = (bf16)b.x; r[5] = (bf16)b.y; r[6] = (bf16)b.z; r[7] = (bf16)b.w;
  return r;
}

// fast silu for bf16 output path: g * rcp(1 + exp(-g))
static __device__ __forceinline__ float fast_silu(float g) {
  return g * __builtin_amdgcn_rcpf(1.0f + __expf(-g));
}

// ---------------- hidden_states fp32 -> bf16 ----------------
__global__ __launch_bounds__(256) void cvt_x_kernel(const float* __restrict__ x,
                                                    bf16* __restrict__ xb) {
  size_t i = ((size_t)blockIdx.x * 256 + threadIdx.x) * 4;
  float4 v = *(const float4*)(x + i);
  bf16* o = xb + i;
  o[0] = (bf16)v.x; o[1] = (bf16)v.y; o[2] = (bf16)v.z; o[3] = (bf16)v.w;
}

// ---------------- routing: softmax + top2 + renorm + compaction ----------------
static __device__ __forceinline__ void top2(const float* s, int& i1, int& i2,
                                            float& w1o, float& w2o) {
  float v1 = s[0]; int a = 0;
  #pragma unroll
  for (int j = 1; j < 8; j++) if (s[j] > v1) { v1 = s[j]; a = j; }
  float v2 = -3.0e38f; int b = 0;
  #pragma unroll
  for (int j = 0; j < 8; j++) { if (j == a) continue; if (s[j] > v2) { v2 = s[j]; b = j; } }
  float e2 = expf(v2 - v1);            // softmax denominator cancels in top-2 renorm
  float inv = 1.0f / (1.0f + e2);
  i1 = a; i2 = b; w1o = inv; w2o = e2 * inv;
}

__global__ __launch_bounds__(256) void route_kernel(const float* __restrict__ score,
                                                    int* __restrict__ cnt,
                                                    int* __restrict__ base,
                                                    int* __restrict__ perm_token,
                                                    float* __restrict__ perm_w,
                                                    int* __restrict__ row_of) {
  __shared__ int scnt[E_NUM];
  __shared__ int sbase[E_NUM];
  const int tid = threadIdx.x;
  if (tid < E_NUM) scnt[tid] = 0;
  __syncthreads();
  for (int t = tid; t < B_TOK; t += 256) {
    float s[8];
    #pragma unroll
    for (int j = 0; j < 8; j++) s[j] = score[t * 8 + j];
    int i1, i2; float wa, wb;
    top2(s, i1, i2, wa, wb);
    atomicAdd(&scnt[i1], 1);
    atomicAdd(&scnt[i2], 1);
  }
  __syncthreads();
  if (tid == 0) {
    int acc = 0;
    for (int e = 0; e < E_NUM; e++) {
      sbase[e] = acc;
      base[e] = acc;
      cnt[e] = scnt[e];
      acc += ((scnt[e] + 127) / 128) * 128;   // 128-padded bases: tile regions never overlap
    }
  }
  __syncthreads();
  if (tid < E_NUM) scnt[tid] = sbase[tid];    // reuse as running offsets
  __syncthreads();
  for (int t = tid; t < B_TOK; t += 256) {
    float s[8];
    #pragma unroll
    for (int j = 0; j < 8; j++) s[j] = score[t * 8 + j];
    int i1, i2; float wa, wb;
    top2(s, i1, i2, wa, wb);
    int r1 = atomicAdd(&scnt[i1], 1);
    int r2 = atomicAdd(&scnt[i2], 1);
    perm_token[r1] = t; perm_w[r1] = wa; row_of[2 * t]     = r1;
    perm_token[r2] = t; perm_w[r2] = wb; row_of[2 * t + 1] = r2;
  }
}

// ---------------- GEMM1: Act[row, i] = silu(x.w1g) * (x.w1u), bf16 WMMA ----------------
__global__ __launch_bounds__(256) void gemm1_kernel(const bf16* __restrict__ Xb,
                                                    const float* __restrict__ w1,
                                                    const int* __restrict__ cnt,
                                                    const int* __restrict__ base,
                                                    const int* __restrict__ perm_token,
                                                    bf16* __restrict__ Act) {
  const int e  = blockIdx.x >> 5;
  const int mt = blockIdx.x & 31;
  const int nc = cnt[e];
  if (mt * 128 >= nc) return;                 // inactive tile -> early exit
  const int rowbase = base[e] + mt * 128;
  const int nb = blockIdx.y * 64;

  __shared__ bf16 As[128][40];                // 80B pitch: conflict-free 16B frag reads
  __shared__ bf16 Wg[64][40];
  __shared__ bf16 Wu[64][40];
  __shared__ int  Ts[128];

  const int tid = threadIdx.x;
  if (tid < 128) {
    int ml = mt * 128 + tid;
    Ts[tid] = (ml < nc) ? perm_token[rowbase + tid] : 0;
  }
  __syncthreads();

  const int wave = tid >> 5, lane = tid & 31;
  const int wm = wave >> 1, wn = wave & 1;    // 4x2 wave grid, 32x32 per wave
  const int half = lane >> 4, lrow = lane & 15;

  const v8f vz = {0.f, 0.f, 0.f, 0.f, 0.f, 0.f, 0.f, 0.f};
  v8f aG[2][2], aU[2][2];
  #pragma unroll
  for (int i = 0; i < 2; i++)
    #pragma unroll
    for (int j = 0; j < 2; j++) { aG[i][j] = vz; aU[i][j] = vz; }

  const int a_row = tid >> 1, a_k = (tid & 1) * 16;
  const int w_row = tid >> 2, w_k = (tid & 3) * 8;
  const bf16*  aptr = Xb + (size_t)Ts[a_row] * D_DIM + a_k;
  const float* gptr = w1 + ((size_t)e * TWO_I + (nb + w_row)) * D_DIM + w_k;
  const float* uptr = gptr + (size_t)I_DIM * D_DIM;

  // software pipeline: registers hold the k-slice being staged while WMMAs run
  bf8_t ra0 = ((const bf8_t*)aptr)[0];
  bf8_t ra1 = ((const bf8_t*)aptr)[1];
  float4 rg0 = *(const float4*)(gptr);
  float4 rg1 = *(const float4*)(gptr + 4);
  float4 ru0 = *(const float4*)(uptr);
  float4 ru1 = *(const float4*)(uptr + 4);

  for (int k0 = 0; k0 < D_DIM; k0 += 32) {
    // stage current slice to LDS
    *(bf8_t*)&As[a_row][a_k]     = ra0;
    *(bf8_t*)&As[a_row][a_k + 8] = ra1;
    *(bf8_t*)&Wg[w_row][w_k] = cvt8(rg0, rg1);
    *(bf8_t*)&Wu[w_row][w_k] = cvt8(ru0, ru1);
    __syncthreads();

    // issue next slice's global loads; they complete behind the WMMAs
    if (k0 + 32 < D_DIM) {
      const int kn = k0 + 32;
      ra0 = ((const bf8_t*)(aptr + kn))[0];
      ra1 = ((const bf8_t*)(aptr + kn))[1];
      rg0 = *(const float4*)(gptr + kn);
      rg1 = *(const float4*)(gptr + kn + 4);
      ru0 = *(const float4*)(uptr + kn);
      ru1 = *(const float4*)(uptr + kn + 4);
    }

    v16bf afrag[2], gfrag[2], ufrag[2];
    #pragma unroll
    for (int i = 0; i < 2; i++) {  // A 16x32 layout: lanes0-15 K[0:8)+K[16:24), lanes16-31 K[8:16)+K[24:32)
      const bf16* p = &As[wm * 32 + i * 16 + lrow][half * 8];
      afrag[i] = SHUF16(*(const bf8_t*)p, *(const bf8_t*)(p + 16));
    }
    #pragma unroll
    for (int j = 0; j < 2; j++) {  // B 32x16 layout: lanes0-15 K[0:8)+K[16:24), lanes16-31 K[8:16)+K[24:32)
      const bf16* pg = &Wg[wn * 32 + j * 16 + lrow][half * 8];
      gfrag[j] = SHUF16(*(const bf8_t*)pg, *(const bf8_t*)(pg + 16));
      const bf16* pu = &Wu[wn * 32 + j * 16 + lrow][half * 8];
      ufrag[j] = SHUF16(*(const bf8_t*)pu, *(const bf8_t*)(pu + 16));
    }
    #pragma unroll
    for (int i = 0; i < 2; i++)
      #pragma unroll
      for (int j = 0; j < 2; j++) {
        aG[i][j] = __builtin_amdgcn_wmma_f32_16x16x32_bf16(false, afrag[i], false, gfrag[j],
                                                           (short)0, aG[i][j], false, false);
        aU[i][j] = __builtin_amdgcn_wmma_f32_16x16x32_bf16(false, afrag[i], false, ufrag[j],
                                                           (short)0, aU[i][j], false, false);
      }
    __syncthreads();   // LDS reads done before next iteration's stores
  }

  // epilogue: silu(g)*u -> bf16; C layout: VGPR r, lanes0-15 M=r,N=lane; lanes16-31 M=r+8
  #pragma unroll
  for (int i = 0; i < 2; i++)
    #pragma unroll
    for (int j = 0; j < 2; j++) {
      const int mbase = wm * 32 + i * 16 + half * 8;
      const int n = nb + wn * 32 + j * 16 + lrow;
      #pragma unroll
      for (int r = 0; r < 8; r++) {
        int mloc = mt * 128 + mbase + r;
        if (mloc < nc) {
          float act = fast_silu(aG[i][j][r]) * aU[i][j][r];
          Act[(size_t)(rowbase + mbase + r) * I_DIM + n] = (bf16)act;
        }
      }
    }
}

// ---------------- GEMM2: P[row, d] = Act[row,:] . w2[e,d,:] ----------------
__global__ __launch_bounds__(256) void gemm2_kernel(const bf16* __restrict__ Act,
                                                    const float* __restrict__ w2,
                                                    const int* __restrict__ cnt,
                                                    const int* __restrict__ base,
                                                    float* __restrict__ P) {
  const int e  = blockIdx.x >> 5;
  const int mt = blockIdx.x & 31;
  const int nc = cnt[e];
  if (mt * 128 >= nc) return;
  const int rowbase = base[e] + mt * 128;
  const int nb = blockIdx.y * 64;

  __shared__ bf16 As[128][40];
  __shared__ bf16 Ws[64][40];

  const int tid = threadIdx.x;
  const int wave = tid >> 5, lane = tid & 31;
  const int wm = wave >> 1, wn = wave & 1;
  const int half = lane >> 4, lrow = lane & 15;

  const v8f vz = {0.f, 0.f, 0.f, 0.f, 0.f, 0.f, 0.f, 0.f};
  v8f acc[2][2];
  #pragma unroll
  for (int i = 0; i < 2; i++)
    #pragma unroll
    for (int j = 0; j < 2; j++) acc[i][j] = vz;

  const int a_row = tid >> 1, a_k = (tid & 1) * 16;
  const int w_row = tid >> 2, w_k = (tid & 3) * 8;
  const bf16*  aptr = Act + (size_t)(rowbase + a_row) * I_DIM + a_k;
  const float* wptr = w2 + ((size_t)e * D_DIM + (nb + w_row)) * I_DIM + w_k;

  bf8_t ra0 = ((const bf8_t*)aptr)[0];
  bf8_t ra1 = ((const bf8_t*)aptr)[1];
  float4 rw0 = *(const float4*)(wptr);
  float4 rw1 = *(const float4*)(wptr + 4);

  for (int k0 = 0; k0 < I_DIM; k0 += 32) {
    *(bf8_t*)&As[a_row][a_k]     = ra0;
    *(bf8_t*)&As[a_row][a_k + 8] = ra1;
    *(bf8_t*)&Ws[w_row][w_k] = cvt8(rw0, rw1);
    __syncthreads();

    if (k0 + 32 < I_DIM) {
      const int kn = k0 + 32;
      ra0 = ((const bf8_t*)(aptr + kn))[0];
      ra1 = ((const bf8_t*)(aptr + kn))[1];
      rw0 = *(const float4*)(wptr + kn);
      rw1 = *(const float4*)(wptr + kn + 4);
    }

    v16bf afrag[2], wfrag[2];
    #pragma unroll
    for (int i = 0; i < 2; i++) {
      const bf16* p = &As[wm * 32 + i * 16 + lrow][half * 8];
      afrag[i] = SHUF16(*(const bf8_t*)p, *(const bf8_t*)(p + 16));
    }
    #pragma unroll
    for (int j = 0; j < 2; j++) {
      const bf16* p = &Ws[wn * 32 + j * 16 + lrow][half * 8];
      wfrag[j] = SHUF16(*(const bf8_t*)p, *(const bf8_t*)(p + 16));
    }
    #pragma unroll
    for (int i = 0; i < 2; i++)
      #pragma unroll
      for (int j = 0; j < 2; j++)
        acc[i][j] = __builtin_amdgcn_wmma_f32_16x16x32_bf16(false, afrag[i], false, wfrag[j],
                                                            (short)0, acc[i][j], false, false);
    __syncthreads();
  }

  #pragma unroll
  for (int i = 0; i < 2; i++)
    #pragma unroll
    for (int j = 0; j < 2; j++) {
      const int mbase = wm * 32 + i * 16 + half * 8;
      const int n = nb + wn * 32 + j * 16 + lrow;
      #pragma unroll
      for (int r = 0; r < 8; r++)
        P[(size_t)(rowbase + mbase + r) * D_DIM + n] = acc[i][j][r];
    }
}

// ---------------- combine: out[t] = w0*P[r0] + w1*P[r1] ----------------
__global__ __launch_bounds__(256) void combine_kernel(const float* __restrict__ P,
                                                      const int* __restrict__ row_of,
                                                      const float* __restrict__ perm_w,
                                                      float* __restrict__ out) {
  const int t = blockIdx.x;
  const int r0 = row_of[2 * t], r1 = row_of[2 * t + 1];
  const float w0 = perm_w[r0], w1v = perm_w[r1];
  const float4* p0 = (const float4*)(P + (size_t)r0 * D_DIM);
  const float4* p1 = (const float4*)(P + (size_t)r1 * D_DIM);
  float4* o = (float4*)(out + (size_t)t * D_DIM);
  for (int c = threadIdx.x; c < D_DIM / 4; c += 256) {
    float4 a = p0[c], b = p1[c], r;
    r.x = w0 * a.x + w1v * b.x;
    r.y = w0 * a.y + w1v * b.y;
    r.z = w0 * a.z + w1v * b.z;
    r.w = w0 * a.w + w1v * b.w;
    o[c] = r;
  }
}

extern "C" void kernel_launch(void* const* d_in, const int* in_sizes, int n_in,
                              void* d_out, int out_size, void* d_ws, size_t ws_size,
                              hipStream_t stream) {
  const float* x     = (const float*)d_in[0];   // [4096, 2048]
  const float* w1    = (const float*)d_in[1];   // [8, 5632, 2048]
  const float* w2    = (const float*)d_in[2];   // [8, 2048, 2816]
  const float* score = (const float*)d_in[3];   // [4096, 8]
  (void)in_sizes; (void)n_in; (void)out_size; (void)ws_size;  // topk fixed at 2

  char* ws = (char*)d_ws;
  int*   cnt        = (int*)(ws + 0);
  int*   base       = (int*)(ws + 64);
  int*   perm_token = (int*)(ws + 256);
  float* perm_w     = (float*)(ws + 256 + 32768);
  int*   row_of     = (int*)(ws + 256 + 65536);
  bf16*  Xb         = (bf16*)(ws + 131072);
  bf16*  Act        = (bf16*)(ws + 131072 + (size_t)B_TOK * D_DIM * 2);
  float* P          = (float*)(ws + 131072 + (size_t)B_TOK * D_DIM * 2
                                           + (size_t)(ROWCAP + 128) * I_DIM * 2);
  // total ws footprint ~139 MB

  cvt_x_kernel<<<(B_TOK * D_DIM) / 1024, 256, 0, stream>>>(x, Xb);
  route_kernel<<<1, 256, 0, stream>>>(score, cnt, base, perm_token, perm_w, row_of);

  dim3 g1(E_NUM * 32, I_DIM / 64);
  gemm1_kernel<<<g1, 256, 0, stream>>>(Xb, w1, cnt, base, perm_token, Act);

  dim3 g2(E_NUM * 32, D_DIM / 64);
  gemm2_kernel<<<g2, 256, 0, stream>>>(Act, w2, cnt, base, P);

  combine_kernel<<<B_TOK, 256, 0, stream>>>(P, row_of, perm_w, (float*)d_out);
}
